// Bottleneck_60919816126994
// MI455X (gfx1250) — matrix operand order repaired and checked by reading, not compiled, not run on hardware
//
#include <hip/hip_runtime.h>

typedef __attribute__((ext_vector_type(16))) _Float16 v16h;
typedef __attribute__((ext_vector_type(8)))  float    v8f;

#define HH_ 28
#define WW_ 28
#define NPOS 784           // 28*28
#define NIMG 64
#define APITCH 34          // LDS row pitch in halves (17 dwords, bank-conflict free)
#define BPITCH 34

// Generic fused implicit-GEMM conv (1x1 or 3x3, pad=KS/2) + BN [+ residual] + ReLU +
// block-floating-point quantization (BLK=32 channels, e=3, m=3 bits).
// Tile: 128 output channels x 64 spatial positions. 128 threads = 4 wave32.
// Wave w owns output channels [32w, 32w+32) of the tile -> one quant block per position.
template<int CIN, int COUT, int KS, typename TIN, bool RESIDUAL, typename TOUT>
__global__ __launch_bounds__(128)
void conv_bn_quant(const TIN* __restrict__ in,
                   const float* __restrict__ wgt,
                   const float* __restrict__ gam, const float* __restrict__ bet,
                   const float* __restrict__ mu,  const float* __restrict__ var,
                   const float* __restrict__ resid,
                   TOUT* __restrict__ out)
{
    __shared__ _Float16 Alds[128 * APITCH];   // weights tile: [co 0..127][k 0..31]
    __shared__ _Float16 Blds[64 * BPITCH];    // act tile, transposed: [p 0..63][k 0..31]

    const int tid  = threadIdx.x;
    const int wave = tid >> 5;
    const int lane = tid & 31;
    const int p0   = blockIdx.x * 64;           // spatial tile base (13 tiles cover 784)
    const int n    = blockIdx.y;                // image
    const int co0  = blockIdx.z * 128;          // output-channel tile base

    v8f acc[2][4];
#pragma unroll
    for (int mt = 0; mt < 2; ++mt)
#pragma unroll
        for (int nt = 0; nt < 4; ++nt)
#pragma unroll
            for (int i = 0; i < 8; ++i) acc[mt][nt][i] = 0.0f;

    const int PAD = KS / 2;
    const int col   = lane & 15;
    const int khalf = lane >> 4;

    for (int k0 = 0; k0 < CIN; k0 += 32) {
        // prefetch hint for the next K-chunk of activations (global_prefetch_b8)
        if (k0 + 32 < CIN && p0 < NPOS) {
            const TIN* pf = in + ((size_t)n * CIN + (k0 + 32 + (tid & 31))) * NPOS + p0;
            __builtin_prefetch(pf, 0, 1);
        }
        for (int tap = 0; tap < KS * KS; ++tap) {
            const int dh = tap / KS - PAD;
            const int dw = tap % KS - PAD;

            __syncthreads();   // protect previous iteration's fragment reads

            // ---- cooperative A load: weights [128 co][32 ci] for this tap, f32->f16 ----
            {
                const int ci    = tid & 31;
                const int corow = tid >> 5;
#pragma unroll
                for (int r = 0; r < 32; ++r) {
                    const int co = corow + (r << 2);            // 0..127
                    const float wv =
                        wgt[((size_t)(co0 + co) * CIN + (k0 + ci)) * (KS * KS) + tap];
                    Alds[co * APITCH + ci] = (_Float16)wv;
                }
            }
            // ---- cooperative B load: shifted input [32 ci][64 p] -> Blds[p][ci], zero-pad OOB ----
            {
                const int pl    = tid & 63;
                const int chalf = tid >> 6;                     // 0..1
                const int p  = p0 + pl;
                const int h  = p / WW_;
                const int w  = p - h * WW_;
                const int hh = h + dh;
                const int ww = w + dw;
                const bool valid = (p < NPOS) && ((unsigned)hh < HH_) && ((unsigned)ww < WW_);
                const size_t base = ((size_t)n * CIN + k0) * NPOS + (size_t)hh * WW_ + ww;
#pragma unroll
                for (int r = 0; r < 16; ++r) {
                    const int cil = chalf * 16 + r;
                    _Float16 v = (_Float16)0.0f;
                    if (valid) v = (_Float16)in[base + (size_t)cil * NPOS];
                    Blds[pl * BPITCH + cil] = v;
                }
            }
            __syncthreads();

            // ---- B fragments: 32(K) x 16(N); lane: col=l&15, K-half=l>>4, vgpr j = K pair ----
            v16h bfrag[4];
#pragma unroll
            for (int nt = 0; nt < 4; ++nt) {
#pragma unroll
                for (int j = 0; j < 8; ++j) {
                    const int kk = khalf * 16 + 2 * j;
                    bfrag[nt][2 * j]     = Blds[(nt * 16 + col) * BPITCH + kk];
                    bfrag[nt][2 * j + 1] = Blds[(nt * 16 + col) * BPITCH + kk + 1];
                }
            }
            // ---- A fragments (16x32) + 8 WMMAs ----
#pragma unroll
            for (int mt = 0; mt < 2; ++mt) {
                v16h afrag;
                const int arow = wave * 32 + mt * 16 + col;
#pragma unroll
                for (int j = 0; j < 8; ++j) {
                    const int kb = khalf * 16 + 2 * j;
                    afrag[2 * j]     = Alds[arow * APITCH + kb];
                    afrag[2 * j + 1] = Alds[arow * APITCH + kb + 1];
                }
#pragma unroll
                for (int nt = 0; nt < 4; ++nt) {
                    acc[mt][nt] = __builtin_amdgcn_wmma_f32_16x16x32_f16(
                        false, afrag, false, bfrag[nt],
                        (short)0, acc[mt][nt], false, false);
                }
            }
        }
    }

    // ---- epilogue: BN [+residual] + ReLU + block-FP quant (32-channel blocks) ----
    // D layout: vgpr r, lane l: M = r + 8*(l>>4), N = l&15.
#pragma unroll
    for (int nt = 0; nt < 4; ++nt) {
        const int p = p0 + nt * 16 + col;
        float vals[2][8];
        float ma = 0.0f;
#pragma unroll
        for (int mt = 0; mt < 2; ++mt) {
#pragma unroll
            for (int r = 0; r < 8; ++r) {
                const int co  = co0 + wave * 32 + mt * 16 + khalf * 8 + r;
                const float inv = gam[co] * __frsqrt_rn(var[co] + 1e-5f);
                float v = acc[mt][nt][r] * inv + (bet[co] - mu[co] * inv);
                if (RESIDUAL) {
                    if (p < NPOS) v += resid[((size_t)n * COUT + co) * NPOS + p];
                }
                v = fmaxf(v, 0.0f);
                vals[mt][r] = v;
                ma = fmaxf(ma, fabsf(v));
            }
        }
        // combine the two row-halves of the 32-channel block (lanes l and l^16)
        ma = fmaxf(ma, __shfl_xor(ma, 16, 32));

        float stepv = 1.0f;
        const bool doq = (ma > 0.0f);
        if (doq) {
            float e = floorf(log2f(ma));
            e = fminf(fmaxf(e, -4.0f), 3.0f);   // 3-bit exponent
            stepv = exp2f(e - 3.0f);            // 3-bit mantissa
        }
        if (p < NPOS) {
#pragma unroll
            for (int mt = 0; mt < 2; ++mt) {
#pragma unroll
                for (int r = 0; r < 8; ++r) {
                    const int co = co0 + wave * 32 + mt * 16 + khalf * 8 + r;
                    float v = vals[mt][r];
                    if (doq) {
                        float q = rintf(v / stepv);
                        q = fminf(fmaxf(q, -16.0f), 15.0f);
                        v = q * stepv;
                    }
                    out[((size_t)n * COUT + co) * NPOS + p] = (TOUT)v;
                }
            }
        }
    }
}

extern "C" void kernel_launch(void* const* d_in, const int* in_sizes, int n_in,
                              void* d_out, int out_size, void* d_ws, size_t ws_size,
                              hipStream_t stream) {
    (void)in_sizes; (void)n_in; (void)out_size; (void)ws_size;
    // setup_inputs() dict order
    const float* x    = (const float*)d_in[0];   // (64,512,28,28)
    const float* w1   = (const float*)d_in[1];   // (128,512,1,1)
    const float* w2   = (const float*)d_in[2];   // (128,128,3,3)
    const float* w3   = (const float*)d_in[3];   // (512,128,1,1)
    const float* g1   = (const float*)d_in[4];
    const float* b1   = (const float*)d_in[5];
    const float* mu1  = (const float*)d_in[6];
    const float* var1 = (const float*)d_in[7];
    const float* g2   = (const float*)d_in[8];
    const float* b2   = (const float*)d_in[9];
    const float* mu2  = (const float*)d_in[10];
    const float* var2 = (const float*)d_in[11];
    const float* g3   = (const float*)d_in[12];
    const float* b3   = (const float*)d_in[13];
    const float* mu3  = (const float*)d_in[14];
    const float* var3 = (const float*)d_in[15];

    // f16 intermediates in workspace (exactly representable: block-FP subset of f16)
    _Float16* act1 = (_Float16*)d_ws;                                  // 64*128*784
    _Float16* act2 = act1 + (size_t)NIMG * 128 * NPOS;                 // 64*128*784

    const dim3 blk(128);
    const dim3 g_a(13, NIMG, 1);   // 13*64 spatial cols cover 784, 128 co
    const dim3 g_c(13, NIMG, 4);   // 512 co -> 4 co-tiles

    // conv1 1x1 512->128 + BN1 + ReLU + quant -> f16
    conv_bn_quant<512, 128, 1, float, false, _Float16>
        <<<g_a, blk, 0, stream>>>(x, w1, g1, b1, mu1, var1, nullptr, act1);
    // conv2 3x3 128->128 pad1 + BN2 + ReLU + quant -> f16
    conv_bn_quant<128, 128, 3, _Float16, false, _Float16>
        <<<g_a, blk, 0, stream>>>(act1, w2, g2, b2, mu2, var2, nullptr, act2);
    // conv3 1x1 128->512 + BN3 + residual + ReLU + quant -> f32 output
    conv_bn_quant<128, 512, 1, _Float16, true, float>
        <<<g_c, blk, 0, stream>>>(act2, w3, g3, b3, mu3, var3, x, (float*)d_out);
}
